// INRRandomGraph_42623255446007
// MI455X (gfx1250) — matrix operand order repaired and checked by reading, not compiled, not run on hardware
//
#include <hip/hip_runtime.h>
#include <stdint.h>

// ---------------- problem constants ----------------
#define WIDTH       64
#define NN          16            // graph nodes
#define NMAT        17            // W1 + 16 gW matrices
#define WAVESZ      32            // gfx1250 is wave32
#define WAVES       4             // waves per workgroup
#define NTHREADS    (WAVES * WAVESZ)
#define ROWS_PER_WG (WAVES * 16)  // 64 batch rows per WG
#define ACT_STRIDE  72            // halves per activation row (pad: 144B, 16B-aligned)
#define SLOT_HALVES (16 * ACT_STRIDE)
#define FRAG_HALVES 512           // one 32x16 f16 B-fragment: 32 lanes * 16 halves
#define MAT_HALVES  4096          // 8 fragments (2 K-blocks x 4 N-tiles) per 64x64 matrix

typedef _Float16 half_t;
typedef __attribute__((ext_vector_type(16))) _Float16 v16h;
typedef __attribute__((ext_vector_type(8)))  _Float16 v8h;
typedef __attribute__((ext_vector_type(8)))  float    v8f;

struct GraphDesc { uint32_t pm[NN]; uint32_t sink_mask; };

// small-constant LDS layout (float offsets)
#define S_WL 0        // Wl 64x8
#define S_BL 512      // bl 64
#define S_WX 576
#define S_WY 640
#define S_WR 704
#define S_B1 768
#define S_GB 832      // gB 16x64
#define S_OW 1856     // outW 3x64
#define S_OB 2048     // outb 3
#define S_SC 2051     // scale
#define S_TOT 2052

// ---------------- weight pre-pack: fp32 -> f16 WMMA B-fragment layout ----------------
// Dense 16-bit B (32x16), matching A: N = lane%16; element j of v16h is
// K = (lane>>4)*8 + j for j<8, K = 16 + (lane>>4)*8 + (j-8) for j>=8.
__global__ void pack_weights(const float* __restrict__ W1,
                             const float* __restrict__ gW,
                             half_t* __restrict__ out) {
  int t = blockIdx.x * blockDim.x + threadIdx.x;
  if (t >= NMAT * MAT_HALVES) return;
  int m    = t / MAT_HALVES;
  int rem  = t % MAT_HALVES;
  int frag = rem / FRAG_HALVES;        // kb*4 + nb
  int lane = (rem >> 4) & 31;
  int j    = rem & 15;
  int kb = frag >> 2, nb = frag & 3;
  int n = nb * 16 + (lane & 15);
  int k = kb * 32 + (j < 8 ? (lane >> 4) * 8 + j : 16 + (lane >> 4) * 8 + (j - 8));
  const float* src = (m == 0) ? W1 : (gW + (size_t)(m - 1) * 4096);
  out[t] = (half_t)src[n * 64 + k];    // B[k][n] = W[n][k]  (x @ W.T)
}

// ---------------- CDNA5 async global->LDS staging (ASYNCcnt-tracked) ----------------
__device__ __forceinline__ void async_copy_b128(const void* gsrc, void* ldst) {
  uint32_t l = (uint32_t)(uintptr_t)ldst;   // generic shared ptr: low 32 bits = LDS offset
  uint64_t a = (uint64_t)(uintptr_t)gsrc;
  asm volatile("global_load_async_to_lds_b128 %0, %1, off"
               :: "v"(l), "v"(a) : "memory");
}
__device__ __forceinline__ void wait_async0() {
#if __has_builtin(__builtin_amdgcn_s_wait_asynccnt)
  __builtin_amdgcn_s_wait_asynccnt(0);
#else
  asm volatile("s_wait_asynccnt 0x0" ::: "memory");
#endif
}

// ---------------- device helpers ----------------
__device__ __forceinline__ float elu_f(float x)      { return x > 0.f ? x : __expf(x) - 1.f; }
__device__ __forceinline__ float softplus_f(float x) { return fmaxf(x, 0.f) + log1pf(__expf(-fabsf(x))); }
__device__ __forceinline__ float gauss_f(float x)    { return __expf(-0.5f * x * x); }
__device__ __forceinline__ float sigmoid_f(float x)  { return 1.f / (1.f + __expf(-x)); }

template<int ACT>
__device__ __forceinline__ float act_t(float v) {
  if constexpr (ACT == 0)      return tanhf(v);
  else if constexpr (ACT == 1) return elu_f(v);
  else if constexpr (ACT == 2) return softplus_f(v);
  else if constexpr (ACT == 3) return __sinf(v);
  else                         return gauss_f(v);
}

// A-fragment (16x32 f16): lane row = lane%16; elems 0..7 = K (lane/16)*8+0..7 (+kb*32),
// elems 8..15 = K 16+(lane/16)*8+0..7.  Two 16B ds_load_b128 per fragment.
__device__ __forceinline__ v16h load_afrag(const half_t* slot, int row, int g2, int kb) {
  const half_t* p = slot + row * ACT_STRIDE + kb * 32 + g2 * 8;
  v8h lo = *(const v8h*)p;
  v8h hi = *(const v8h*)(p + 16);
  return __builtin_shufflevector(lo, hi, 0,1,2,3,4,5,6,7,8,9,10,11,12,13,14,15);
}

__device__ __forceinline__ v16h load_bfrag(const half_t* sW, int m, int kb, int nt, int lane) {
  const half_t* p = sW + ((size_t)((m * 2 + kb) * 4 + nt) * 32 + lane) * 16;
  v8h lo = *(const v8h*)p;
  v8h hi = *(const v8h*)(p + 8);
  return __builtin_shufflevector(lo, hi, 0,1,2,3,4,5,6,7,8,9,10,11,12,13,14,15);
}

// One 16x64 * 64x64 matmul tile on WMMA, then bias + activation + f16 store to LDS slot.
template<int ACT>
__device__ __forceinline__ void mm_act_store(const half_t* sW, int m, v16h a0, v16h a1,
                                             half_t* dst, int lane, const float* bias) {
  int row0 = (lane >> 4) * 8;
  int coll = lane & 15;
#pragma unroll
  for (int nt = 0; nt < 4; ++nt) {
    v16h b0 = load_bfrag(sW, m, 0, nt, lane);
    v16h b1 = load_bfrag(sW, m, 1, nt, lane);
    v8f acc = {};
    acc = __builtin_amdgcn_wmma_f32_16x16x32_f16(false, a0, false, b0, (short)0, acc, false, false);
    acc = __builtin_amdgcn_wmma_f32_16x16x32_f16(false, a1, false, b1, (short)0, acc, false, false);
    int col = nt * 16 + coll;
    float bb = bias[col];
#pragma unroll
    for (int p = 0; p < 8; ++p) {
      float v = act_t<ACT>(acc[p] + bb);
      dst[(row0 + p) * ACT_STRIDE + col] = (half_t)v;
    }
  }
}

// One graph node, fully static (J compile-time => activation branch-free, 8 static WMMA).
template<int J>
__device__ __forceinline__ void node_step(const half_t* sW, half_t* actW, const half_t* fslot,
                                          const float* sSmall, const GraphDesc& g,
                                          int row, int g2, int lane) {
  uint32_t pm = g.pm[J];
  v16h a0, a1;
  if (pm == 0) {                     // source node: input is f
    a0 = load_afrag(fslot, row, g2, 0);
    a1 = load_afrag(fslot, row, g2, 1);
  } else {                           // sum of predecessor h tiles (packed f16 adds)
    int first = 1;
    a0 = (v16h)(_Float16)0; a1 = (v16h)(_Float16)0;
    for (uint32_t mbits = pm; mbits; mbits &= mbits - 1) {
      int i = __ffs(mbits) - 1;
      const half_t* s = actW + i * SLOT_HALVES;
      v16h t0 = load_afrag(s, row, g2, 0);
      v16h t1 = load_afrag(s, row, g2, 1);
      if (first) { a0 = t0; a1 = t1; first = 0; }
      else       { a0 += t0; a1 += t1; }           // v_pk_add_f16
    }
  }
  mm_act_store<J % 5>(sW, 1 + J, a0, a1, actW + J * SLOT_HALVES, lane,
                      sSmall + S_GB + J * 64);
}

// ---------------- main kernel ----------------
__global__ __launch_bounds__(NTHREADS, 1)
void inr_graph_kernel(const float* __restrict__ inputs, const float* __restrict__ latents,
                      const float* __restrict__ Wl, const float* __restrict__ bl,
                      const float* __restrict__ Wx, const float* __restrict__ Wy,
                      const float* __restrict__ Wr, const float* __restrict__ b1,
                      const float* __restrict__ gB, const float* __restrict__ outW,
                      const float* __restrict__ outb, const float* __restrict__ scale,
                      const half_t* __restrict__ wpack, float* __restrict__ out,
                      GraphDesc g) {
  __attribute__((aligned(16))) __shared__ half_t sW[NMAT * MAT_HALVES];            // 136 KB
  __attribute__((aligned(16))) __shared__ half_t sAct[WAVES * NMAT * SLOT_HALVES]; // 153 KB
  __shared__ float sSmall[S_TOT];                                                  // 8 KB

  const int tid  = threadIdx.x;
  const int wave = tid >> 5;
  const int lane = tid & 31;

  // ---- stage packed f16 weights into LDS with async DMA (no VGPR round-trip) ----
  {
    const char* src = (const char*)wpack;
    char*       dst = (char*)sW;
    const int bytes = NMAT * MAT_HALVES * 2;                 // 139264 = 128*16*68
    for (int off = tid * 16; off < bytes; off += NTHREADS * 16)
      async_copy_b128(src + off, dst + off);
    // small fp32 constants via regular loads
    for (int i = tid; i < 512;  i += NTHREADS) sSmall[S_WL + i] = Wl[i];
    for (int i = tid; i < 64;   i += NTHREADS) {
      sSmall[S_BL + i] = bl[i];  sSmall[S_WX + i] = Wx[i];
      sSmall[S_WY + i] = Wy[i];  sSmall[S_WR + i] = Wr[i];
      sSmall[S_B1 + i] = b1[i];
    }
    for (int i = tid; i < NN * 64; i += NTHREADS) sSmall[S_GB + i] = gB[i];
    for (int i = tid; i < 192;     i += NTHREADS) sSmall[S_OW + i] = outW[i];
    if (tid < 3) sSmall[S_OB + tid] = outb[tid];
    if (tid == 3) sSmall[S_SC] = scale[0];
    wait_async0();
  }
  __syncthreads();

  half_t* actW  = sAct + (size_t)wave * NMAT * SLOT_HALVES; // wave-private slots
  half_t* fslot = actW + NN * SLOT_HALVES;                  // slot 16 = f
  const int row  = lane & 15;
  const int g2   = lane >> 4;
  const int grow = blockIdx.x * ROWS_PER_WG + wave * 16 + row;

  // ---- head: f_pre = gaussian(tanh(x Wx) + softplus(y Wy) + elu(r Wr) + tanh(lat@Wl.T + bl))
  {
    float x = inputs[grow * 3 + 0];
    float y = inputs[grow * 3 + 1];
    float r = inputs[grow * 3 + 2];
    float lat[8];
#pragma unroll
    for (int i = 0; i < 8; ++i) lat[i] = latents[grow * 8 + i];
    for (int c = 0; c < 32; ++c) {
      int col = g2 * 32 + c;
      float lt = sSmall[S_BL + col];
#pragma unroll
      for (int i = 0; i < 8; ++i) lt += lat[i] * sSmall[S_WL + col * 8 + i];
      float v = tanhf(x * sSmall[S_WX + col]) + softplus_f(y * sSmall[S_WY + col]) +
                elu_f(r * sSmall[S_WR + col]) + tanhf(lt);
      fslot[row * ACT_STRIDE + col] = (half_t)gauss_f(v);
    }
  }

  // ---- f = sin(f_pre @ W1.T + b1)  (matrix 0; writes back into slot 16)
  {
    v16h a0 = load_afrag(fslot, row, g2, 0);
    v16h a1 = load_afrag(fslot, row, g2, 1);
    mm_act_store<3>(sW, 0, a0, a1, fslot, lane, sSmall + S_B1);
  }

  // ---- graph nodes, statically unrolled: 16 x 8 WMMA, branch-free activations
  node_step< 0>(sW, actW, fslot, sSmall, g, row, g2, lane);
  node_step< 1>(sW, actW, fslot, sSmall, g, row, g2, lane);
  node_step< 2>(sW, actW, fslot, sSmall, g, row, g2, lane);
  node_step< 3>(sW, actW, fslot, sSmall, g, row, g2, lane);
  node_step< 4>(sW, actW, fslot, sSmall, g, row, g2, lane);
  node_step< 5>(sW, actW, fslot, sSmall, g, row, g2, lane);
  node_step< 6>(sW, actW, fslot, sSmall, g, row, g2, lane);
  node_step< 7>(sW, actW, fslot, sSmall, g, row, g2, lane);
  node_step< 8>(sW, actW, fslot, sSmall, g, row, g2, lane);
  node_step< 9>(sW, actW, fslot, sSmall, g, row, g2, lane);
  node_step<10>(sW, actW, fslot, sSmall, g, row, g2, lane);
  node_step<11>(sW, actW, fslot, sSmall, g, row, g2, lane);
  node_step<12>(sW, actW, fslot, sSmall, g, row, g2, lane);
  node_step<13>(sW, actW, fslot, sSmall, g, row, g2, lane);
  node_step<14>(sW, actW, fslot, sSmall, g, row, g2, lane);
  node_step<15>(sW, actW, fslot, sSmall, g, row, g2, lane);

  // ---- output: sigmoid((sum_sinks h @ outW.T + outb) * scale)
  if (lane < 16) {
    float s0 = 0.f, s1 = 0.f, s2 = 0.f;
    for (int k = 0; k < 64; ++k) {
      float a = 0.f;
      for (uint32_t mbits = g.sink_mask; mbits; mbits &= mbits - 1) {
        int i = __ffs(mbits) - 1;
        a += (float)actW[i * SLOT_HALVES + row * ACT_STRIDE + k];
      }
      s0 += a * sSmall[S_OW + 0 * 64 + k];
      s1 += a * sSmall[S_OW + 1 * 64 + k];
      s2 += a * sSmall[S_OW + 2 * 64 + k];
    }
    float sc = sSmall[S_SC];
    out[grow * 3 + 0] = sigmoid_f((s0 + sSmall[S_OB + 0]) * sc);
    out[grow * 3 + 1] = sigmoid_f((s1 + sSmall[S_OB + 1]) * sc);
    out[grow * 3 + 2] = sigmoid_f((s2 + sSmall[S_OB + 2]) * sc);
  }
}

// ---------------- host: numpy-legacy MT19937 replica + graph build ----------------
namespace {
struct NpMT {
  uint32_t key[624]; int pos;
  void seed(uint32_t s) {               // numpy rk_seed
    for (int i = 0; i < 624; ++i) {
      key[i] = s;
      s = 1812433253u * (s ^ (s >> 30)) + (uint32_t)i + 1u;
    }
    pos = 624;
  }
  uint32_t next() {
    if (pos >= 624) {
      for (int i = 0; i < 624; ++i) {
        uint32_t y = (key[i] & 0x80000000u) | (key[(i + 1) % 624] & 0x7fffffffu);
        key[i] = key[(i + 397) % 624] ^ (y >> 1) ^ ((y & 1u) ? 0x9908b0dfu : 0u);
      }
      pos = 0;
    }
    uint32_t y = key[pos++];
    y ^= y >> 11; y ^= (y << 7) & 0x9d2c5680u; y ^= (y << 15) & 0xefc60000u; y ^= y >> 18;
    return y;
  }
  double rand_d() {                      // rk_double: 53-bit from two draws
    uint32_t a = next() >> 5, b = next() >> 6;
    return (a * 67108864.0 + b) / 9007199254740992.0;
  }
  uint32_t randint(uint32_t n) {         // legacy masked rejection on 32-bit draws
    if (n <= 1) return 0;
    uint32_t max = n - 1, mask = max, v;
    mask |= mask >> 1; mask |= mask >> 2; mask |= mask >> 4; mask |= mask >> 8; mask |= mask >> 16;
    while ((v = next() & mask) > max) {}
    return v;
  }
};

GraphDesc build_graph_host() {
  NpMT rng; rng.seed(0);
  bool edge[NN][NN] = {};
  for (int i = 0; i < NN; ++i) {
    for (int d = 1; d <= 2; ++d) {       // k=4 -> d in {1,2}
      int j = (i + d) % NN;
      if (rng.rand_d() < 0.75) j = (int)rng.randint(NN);
      int a = i < j ? i : j, b = i < j ? j : i;
      if (a != b) edge[a][b] = true;
    }
  }
  uint32_t pred[NN] = {};
  for (int a = 0; a < NN; ++a)
    for (int b = 0; b < NN; ++b)
      if (edge[a][b]) pred[b] |= 1u << a;
  for (int j = 1; j < NN; ++j)
    if (!pred[j]) { int a = (int)rng.randint((uint32_t)j); pred[j] |= 1u << a; edge[a][j] = true; }
  uint32_t succ[NN] = {};
  for (int a = 0; a < NN; ++a)
    for (int b = 0; b < NN; ++b)
      if (edge[a][b]) succ[a] |= 1u << b;
  GraphDesc g;
  g.sink_mask = 0;
  for (int j = 0; j < NN; ++j) {
    g.pm[j] = pred[j];                   // 0 => source => input is f
    if (succ[j] == 0) g.sink_mask |= 1u << j;
  }
  return g;
}
} // namespace

extern "C" void kernel_launch(void* const* d_in, const int* in_sizes, int n_in,
                              void* d_out, int out_size, void* d_ws, size_t ws_size,
                              hipStream_t stream) {
  (void)n_in; (void)out_size; (void)ws_size;
  const float* inputs  = (const float*)d_in[0];
  const float* latents = (const float*)d_in[1];
  const float* Wl      = (const float*)d_in[2];
  const float* bl      = (const float*)d_in[3];
  const float* Wx      = (const float*)d_in[4];
  const float* Wy      = (const float*)d_in[5];
  const float* Wr      = (const float*)d_in[6];
  const float* W1      = (const float*)d_in[7];
  const float* b1      = (const float*)d_in[8];
  const float* gW      = (const float*)d_in[9];
  const float* gB      = (const float*)d_in[10];
  const float* outW    = (const float*)d_in[11];
  const float* outb    = (const float*)d_in[12];
  const float* scale   = (const float*)d_in[13];
  float* out = (float*)d_out;
  half_t* wpack = (half_t*)d_ws;         // 17*4096 halves = 136 KB scratch

  const int rows = in_sizes[0] / 3;      // B = 131072

  {
    int total = NMAT * MAT_HALVES;
    pack_weights<<<(total + 255) / 256, 256, 0, stream>>>(W1, gW, wpack);
  }

  GraphDesc g = build_graph_host();      // deterministic, recomputed every call

  inr_graph_kernel<<<rows / ROWS_PER_WG, NTHREADS, 0, stream>>>(
      inputs, latents, Wl, bl, Wx, Wy, Wr, b1, gB, outW, outb, scale,
      wpack, out, g);
}
